// TDGCNCell_58926951301241
// MI455X (gfx1250) — matrix operand order repaired and checked by reading, not compiled, not run on hardware
//
#include <hip/hip_runtime.h>

#define NN 4096
#define TT 64
#define HH 128

typedef __bf16 bf16;
typedef __attribute__((ext_vector_type(16))) __bf16 v16bf;
typedef __attribute__((ext_vector_type(8)))  __bf16 v8bf;
typedef __attribute__((ext_vector_type(8)))  float  v8f;

// 128^-0.25 : pre-scale applied to h_proj so that Q.K^T carries 1/sqrt(H)
#define QSCALE 0.29730177875068026f

#define V8F_ZERO ((v8f){0.f, 0.f, 0.f, 0.f, 0.f, 0.f, 0.f, 0.f})

__device__ __forceinline__ v8f wmma_bf16(v16bf a, v16bf b, v8f c) {
  return __builtin_amdgcn_wmma_f32_16x16x32_bf16(false, a, false, b,
                                                 (short)0, c, false, false);
}

// A-fragment (16x32, 16-bit): lane holds row M=lane%16;
// elements 0..7 -> K = 8*hi + e, elements 8..15 -> K = 16 + 8*hi + (e-8)
__device__ __forceinline__ v16bf load_a_frag(const bf16* p, int kOffA) {
  v8bf lo = *(const v8bf*)(p + kOffA);
  v8bf hi = *(const v8bf*)(p + 16 + kOffA);
  return __builtin_shufflevector(lo, hi, 0,1,2,3,4,5,6,7,8,9,10,11,12,13,14,15);
}

// Load the 8 K A-fragments (two 16-key tiles x 4 K-chunks) for a 32-key chunk.
__device__ __forceinline__ void load_afrags(v16bf F[8], const bf16* __restrict__ qk,
                                            int kBase, int lr, int kOffA) {
  const bf16* kr0 = qk + (size_t)(kBase + lr) * HH;
  const bf16* kr1 = kr0 + 16 * HH;
#pragma unroll
  for (int c = 0; c < 4; ++c) {
    F[c]     = load_a_frag(kr0 + 32 * c, kOffA);
    F[4 + c] = load_a_frag(kr1 + 32 * c, kOffA);
  }
}

// Online-softmax step for one 16-query set over a 32-key chunk.
// sA/sB are the two 16x16 transposed score tiles (lane = query, vgpr r =
// key r + 8*hi). Updates (m_run, l_run), rescales acc[8], returns the
// probability A-fragment P (16 queries x 32 keys) ready for the PV WMMA.
__device__ __forceinline__ v16bf softmax_step(const v8f& sA, const v8f& sB,
                                              float& m_run, float& l_run,
                                              v8f acc[8], int hi8) {
  float tmax = sA[0];
#pragma unroll
  for (int r = 1; r < 8; ++r) tmax = fmaxf(tmax, sA[r]);
#pragma unroll
  for (int r = 0; r < 8; ++r) tmax = fmaxf(tmax, sB[r]);
  tmax = fmaxf(tmax, __shfl_xor(tmax, 16, 32));
  const float m_new = fmaxf(m_run, tmax);
  const float ef    = __expf(m_run - m_new);
  float p0[8], p1[8], ts = 0.f;
#pragma unroll
  for (int r = 0; r < 8; ++r) { p0[r] = __expf(sA[r] - m_new); ts += p0[r]; }
#pragma unroll
  for (int r = 0; r < 8; ++r) { p1[r] = __expf(sB[r] - m_new); ts += p1[r]; }
  ts += __shfl_xor(ts, 16, 32);
  l_run = l_run * ef + ts;
  m_run = m_new;
  // factor for acc row M = r + 8*hi lives at lane M (replicated at M+16)
  float fr[8];
#pragma unroll
  for (int r = 0; r < 8; ++r) fr[r] = __shfl(ef, r + hi8, 32);
#pragma unroll
  for (int f = 0; f < 8; ++f)
#pragma unroll
    for (int r = 0; r < 8; ++r) acc[f][r] *= fr[r];
  // score C-layout == P A-fragment layout: e<8 -> p0[e], e>=8 -> p1[e-8]
  v16bf P;
#pragma unroll
  for (int r = 0; r < 8; ++r) { P[r] = (bf16)p0[r]; P[8 + r] = (bf16)p1[r]; }
  return P;
}

// One 32-key chunk: V loads issued first (consumed 16 WMMAs + softmax later),
// score WMMAs from pre-loaded Fcur, next chunk's A-fragments loaded into
// Fnext (latency hidden behind softmax + PV), then PV WMMAs.
__device__ __forceinline__ void process_chunk(
    int kA, int kNext, const bf16* __restrict__ qk, const bf16* __restrict__ vt,
    const v16bf Fcur[8], v16bf Fnext[8],
    const v16bf Bq0[4], const v16bf Bq1[4],
    v8f acc0[8], v8f acc1[8],
    float& m0, float& l0, float& m1, float& l1,
    int lr, int hi8, int kOffA, int kOffB)
{
  // 1. V B-fragments for this chunk (one clause, long in-flight window)
  v16bf Fv[8];
#pragma unroll
  for (int f = 0; f < 8; ++f)
    Fv[f] = load_a_frag(vt + (size_t)(f * 16 + lr) * NN + kA, kOffB);

  // 2. score WMMAs: S~ = K_tile . Q^T, 2 key tiles x 2 query sets
  v8f sA0 = V8F_ZERO, sA1 = V8F_ZERO, sB0 = V8F_ZERO, sB1 = V8F_ZERO;
#pragma unroll
  for (int c = 0; c < 4; ++c) {
    sA0 = wmma_bf16(Fcur[c],     Bq0[c], sA0);
    sA1 = wmma_bf16(Fcur[c],     Bq1[c], sA1);
    sB0 = wmma_bf16(Fcur[4 + c], Bq0[c], sB0);
    sB1 = wmma_bf16(Fcur[4 + c], Bq1[c], sB1);
  }

  // 3. software prefetch of next chunk's A-fragments
  load_afrags(Fnext, qk, kNext, lr, kOffA);

  // 4. online softmax (VALU/TRANS co-executes with in-flight loads)
  v16bf P0 = softmax_step(sA0, sB0, m0, l0, acc0, hi8);
  v16bf P1 = softmax_step(sA1, sB1, m1, l1, acc1, hi8);

  // 5. msg += P @ V : each V B-fragment feeds both query sets
#pragma unroll
  for (int f = 0; f < 8; ++f) {
    acc0[f] = wmma_bf16(P0, Fv[f], acc0[f]);
    acc1[f] = wmma_bf16(P1, Fv[f], acc1[f]);
  }
}

// -------------------------------------------------------------------------
// Flash attention:  msg = softmax(h_proj h_proj^T * s) @ h_in
// Block = 128 threads (4 waves). Block owns 32 query rows (two 16-row
// fragment sets); each wave streams 1024 keys (split-K=4), 32 WMMAs per
// 32 b128 loads, double-buffered A-fragments. Waves 1..3 stage partials
// in LDS; wave 0 merges against its registers.
// -------------------------------------------------------------------------
__global__ __launch_bounds__(128) void attn_kernel(
    const bf16* __restrict__ qk,   // [NN][HH] bf16, pre-scaled h_proj
    const bf16* __restrict__ vt,   // [HH][NN] bf16, h_in transposed
    float* __restrict__ msg)       // [NN][HH] f32
{
  __shared__ float s_acc[3][128][32];   // 48 KB
  __shared__ float s_lm[4][2][32];
  __shared__ float s_ll[4][2][32];

  const int lane  = threadIdx.x & 31;
  const int wv    = threadIdx.x >> 5;
  const int qBase = blockIdx.x * 32;
  const int lr    = lane & 15;
  const int hi    = lane >> 4;      // 0 or 1 (half-wave)
  const int hi8   = hi * 8;
  const int kOffA = hi * 8;         // A-fragment interleave
  const int kOffB = hi * 8;         // B-fragment interleave (matches A)

  // Preload 32 queries as two B-fragment sets: B[k][n] = Q[n][k]
  v16bf Bq0[4], Bq1[4];
  {
    const bf16* q0 = qk + (size_t)(qBase + lr) * HH;
    const bf16* q1 = q0 + 16 * HH;
#pragma unroll
    for (int c = 0; c < 4; ++c) {
      Bq0[c] = load_a_frag(q0 + 32 * c, kOffB);
      Bq1[c] = load_a_frag(q1 + 32 * c, kOffB);
    }
  }

  v8f acc0[8], acc1[8];
#pragma unroll
  for (int f = 0; f < 8; ++f) { acc0[f] = V8F_ZERO; acc1[f] = V8F_ZERO; }
  float m0 = -1e30f, l0 = 0.0f, m1 = -1e30f, l1 = 0.0f;

  const int KSEG = NN / 4;          // keys per wave
  const int k0 = wv * KSEG;

  v16bf Fa[8], Fb[8];
  load_afrags(Fa, qk, k0, lr, kOffA);

  for (int kb = 0; kb < KSEG; kb += 64) {
    const int kA = k0 + kb;
    const int kB = kA + 32;
    const int kN = (kb + 64 < KSEG) ? (kA + 64) : k0;  // wrap: dummy preload
    process_chunk(kA, kB, qk, vt, Fa, Fb, Bq0, Bq1, acc0, acc1,
                  m0, l0, m1, l1, lr, hi8, kOffA, kOffB);
    process_chunk(kB, kN, qk, vt, Fb, Fa, Bq0, Bq1, acc0, acc1,
                  m0, l0, m1, l1, lr, hi8, kOffA, kOffB);
  }

  // Publish stats (all waves) and partial acc (waves 1..3).
  s_lm[wv][0][lane] = m0; s_lm[wv][1][lane] = m1;
  s_ll[wv][0][lane] = l0; s_ll[wv][1][lane] = l1;
  if (wv > 0) {
#pragma unroll
    for (int f = 0; f < 8; ++f)
#pragma unroll
      for (int r = 0; r < 8; ++r) {
        s_acc[wv - 1][f * 8 + r][lane]      = acc0[f][r];
        s_acc[wv - 1][64 + f * 8 + r][lane] = acc1[f][r];
      }
  }
  __syncthreads();

  if (wv == 0) {
#pragma unroll
    for (int set = 0; set < 2; ++set) {
#pragma unroll
      for (int r = 0; r < 8; ++r) {
        const int q = r + hi8;           // query row this vgpr slot holds
        const float mw0 = s_lm[0][set][q], mw1 = s_lm[1][set][q];
        const float mw2 = s_lm[2][set][q], mw3 = s_lm[3][set][q];
        const float ms = fmaxf(fmaxf(mw0, mw1), fmaxf(mw2, mw3));
        const float e0 = __expf(mw0 - ms), e1 = __expf(mw1 - ms);
        const float e2 = __expf(mw2 - ms), e3 = __expf(mw3 - ms);
        const float ls = s_ll[0][set][q] * e0 + s_ll[1][set][q] * e1 +
                         s_ll[2][set][q] * e2 + s_ll[3][set][q] * e3;
        const float inv = 1.0f / ls;
        const int row = qBase + set * 16 + q;
#pragma unroll
        for (int f = 0; f < 8; ++f) {
          const int idx = set * 64 + f * 8 + r;
          const float own = set ? acc1[f][r] : acc0[f][r];
          float v = own * e0 + s_acc[0][idx][lane] * e1 +
                    s_acc[1][idx][lane] * e2 + s_acc[2][idx][lane] * e3;
          msg[(size_t)row * HH + f * 16 + lr] = v * inv;
        }
      }
    }
  }
}

// -------------------------------------------------------------------------
// Per-node update: h_new = relu(msg@Wg + bg + h_in); pred; and next step's
// h_in / h_inT(bf16) / h_proj(bf16, pre-scaled). Block = 128 thr = one node.
// -------------------------------------------------------------------------
__global__ __launch_bounds__(128) void update_kernel(
    const float* __restrict__ msg, const float* __restrict__ h_in,
    const float* __restrict__ Wg, const float* __restrict__ bg,
    const float* __restrict__ Wo, const float* __restrict__ bo,
    const float* __restrict__ Wp, const float* __restrict__ bp,
    const float* __restrict__ Wi, const float* __restrict__ bi,
    const float* __restrict__ x_seq, const float* __restrict__ m_seq,
    float* __restrict__ out, float* __restrict__ h_in_next,
    bf16* __restrict__ h_inT_next, bf16* __restrict__ qk_next,
    int t, int last)
{
  __shared__ float sm[HH];
  __shared__ float sh[HH];
  __shared__ float sred[4];
  const int i = blockIdx.x, j = threadIdx.x;

  sm[j] = msg[(size_t)i * HH + j];
  __syncthreads();

  float a = bg[j];
#pragma unroll 8
  for (int k = 0; k < HH; ++k) a += sm[k] * Wg[k * HH + j];
  const float hn = fmaxf(a + h_in[(size_t)i * HH + j], 0.f);
  sh[j] = hn;
  __syncthreads();

  // pred = h_new . W_out + b_out
  float pp = hn * Wo[j];
#pragma unroll
  for (int off = 16; off > 0; off >>= 1) pp += __shfl_down(pp, off, 32);
  if ((j & 31) == 0) sred[j >> 5] = pp;
  __syncthreads();
  if (j == 0) out[(size_t)i * TT + t] = sred[0] + sred[1] + sred[2] + sred[3] + bo[0];

  if (!last) {
    // next-step projection (Q/K), bf16, pre-scaled by H^-1/4
    float pr = bp[j];
#pragma unroll 8
    for (int k = 0; k < HH; ++k) pr += sh[k] * Wp[k * HH + j];
    qk_next[(size_t)i * HH + j] = (bf16)(pr * QSCALE);

    // next-step h_in (V), f32 row-major + bf16 transposed
    const float xv = x_seq[(size_t)i * TT + t + 1];
    const float mv = m_seq[(size_t)i * TT + t + 1];
    const float hv = fmaxf(xv * Wi[j] + mv * Wi[HH + j] + bi[j], 0.f);
    h_in_next[(size_t)i * HH + j] = hv;
    h_inT_next[(size_t)j * NN + i] = (bf16)hv;
  }
}

// t = 0 bootstrap: h0 == 0 so h_proj_0 = b_proj.
__global__ __launch_bounds__(128) void init_kernel(
    const float* __restrict__ x_seq, const float* __restrict__ m_seq,
    const float* __restrict__ Wi, const float* __restrict__ bi,
    const float* __restrict__ bp,
    float* __restrict__ h_in, bf16* __restrict__ h_inT, bf16* __restrict__ qk)
{
  const int i = blockIdx.x, j = threadIdx.x;
  const float xv = x_seq[(size_t)i * TT];
  const float mv = m_seq[(size_t)i * TT];
  const float hv = fmaxf(xv * Wi[j] + mv * Wi[HH + j] + bi[j], 0.f);
  h_in[(size_t)i * HH + j] = hv;
  h_inT[(size_t)j * NN + i] = (bf16)hv;
  qk[(size_t)i * HH + j] = (bf16)(bp[j] * QSCALE);
}

extern "C" void kernel_launch(void* const* d_in, const int* in_sizes, int n_in,
                              void* d_out, int out_size, void* d_ws, size_t ws_size,
                              hipStream_t stream) {
  (void)in_sizes; (void)n_in; (void)out_size; (void)ws_size;
  const float* x_seq = (const float*)d_in[0];
  const float* m_seq = (const float*)d_in[1];
  const float* W_in  = (const float*)d_in[2];
  const float* b_in  = (const float*)d_in[3];
  const float* W_prj = (const float*)d_in[4];
  const float* b_prj = (const float*)d_in[5];
  const float* W_gcn = (const float*)d_in[6];
  const float* b_gcn = (const float*)d_in[7];
  const float* W_out = (const float*)d_in[8];
  const float* b_out = (const float*)d_in[9];
  float* out = (float*)d_out;

  char* ws = (char*)d_ws;
  float* msg   = (float*)(ws);                                   // 2 MB
  float* h_in  = (float*)(ws + (size_t)NN * HH * 4);             // 2 MB
  bf16*  h_inT = (bf16*)(ws + (size_t)NN * HH * 8);              // 1 MB
  bf16*  qk    = (bf16*)(ws + (size_t)NN * HH * 8 + (size_t)NN * HH * 2); // 1 MB

  init_kernel<<<NN, HH, 0, stream>>>(x_seq, m_seq, W_in, b_in, b_prj,
                                     h_in, h_inT, qk);
  for (int t = 0; t < TT; ++t) {
    attn_kernel<<<NN / 32, 128, 0, stream>>>(qk, h_inT, msg);
    update_kernel<<<NN, HH, 0, stream>>>(msg, h_in, W_gcn, b_gcn, W_out, b_out,
                                         W_prj, b_prj, W_in, b_in, x_seq, m_seq,
                                         out, h_in, h_inT, qk, t,
                                         (t == TT - 1) ? 1 : 0);
  }
}
